// DilatedRNN_11793980195433
// MI455X (gfx1250) — matrix operand order repaired and checked by reading, not compiled, not run on hardware
//
#include <hip/hip_runtime.h>
#include <hip/hip_bf16.h>

// Dilated RNN (8 layers, dil 1..128, H=128, B=256, T=784) for MI455X (gfx1250).
//
// Per layer:
//   (1) U = Hprev @ Wih^T + (bih+bhh)  -> parallel bf16 WMMA GEMM over T*B rows
//   (2) scan: h_t = tanh(U_t + h_{t-rate} @ Whh^T)  -> persistent workgroups,
//       running h tile (16x128) in LDS, recurrent-weight B-operands pre-packed
//       into the exact WMMA register image and held in VGPRs across the loop.
//
// ws layout: U bf16 [T*B][128] | Ha bf16 | Hb bf16 | whhpack[8][16384] bf16 |
//            wihpack[7][16384] bf16   (~154.6 MB total)

#define T_STEPS 784
#define BATCH   256
#define HID     128

typedef __attribute__((ext_vector_type(16))) __bf16 v16bf;
typedef __attribute__((ext_vector_type(8)))  __bf16 bf16x8;
typedef __attribute__((ext_vector_type(8)))  float  v8f;

// Native f32 -> bf16 convert (RNE): let the backend emit the hw cvt.
__device__ __forceinline__ unsigned short f2bf(float f) {
    __bf16 b = (__bf16)f;
    unsigned short u;
    __builtin_memcpy(&u, &b, 2);
    return u;
}
__device__ __forceinline__ float bf2f(unsigned short h) {
    union { unsigned u; float f; } a; a.u = ((unsigned)h) << 16;
    return a.f;
}
__device__ __forceinline__ float fast_tanh(float x) {
#if __has_builtin(__builtin_amdgcn_tanhf)
    return __builtin_amdgcn_tanhf(x);   // native v_tanh_f32 (TRANS32)
#else
    return tanhf(x);
#endif
}

// ---------------------------------------------------------------------------
// One-shot weight packer: write each 128x128 weight matrix as the WMMA
// B-operand register image, so a lane's 16 bf16 values per 32-K slice are
// contiguous (one v16bf load, no LDS staging, no element packing).
//   pack[((wave*4+s)*32 + lane)*16 + e] = bf16(W[n*128 + k])
//   n = wave*16 + (lane&15),  k = s*32 + (lane>>4)*8 + (e&7) + (e>>3)*16
// Matrices 0..7: Whh (layer 0 uses Whh0); 8..14: Wih (layers 1..7).
// ---------------------------------------------------------------------------
__global__ __launch_bounds__(256) void drnn_pack(
    const float* __restrict__ Whh0, const float* __restrict__ Wih,
    const float* __restrict__ Whh,
    unsigned short* __restrict__ whhp, unsigned short* __restrict__ wihp)
{
    int i = blockIdx.x * 256 + threadIdx.x;
    if (i >= 15 * 16384) return;
    int m = i >> 14;
    int e = i & 16383;
    int el   = e & 15;
    int lane = (e >> 4) & 31;
    int s    = (e >> 9) & 3;
    int wave = e >> 11;
    int n = wave * 16 + (lane & 15);
    int k = s * 32 + (lane >> 4) * 8 + (el & 7) + (el >> 3) * 16;
    const float* src;
    unsigned short* dst;
    if (m < 8) {
        src = (m == 0) ? Whh0 : (Whh + (size_t)(m - 1) * 16384);
        dst = whhp + (size_t)m * 16384;
    } else {
        src = Wih + (size_t)(m - 8) * 16384;
        dst = wihp + (size_t)(m - 8) * 16384;
    }
    dst[e] = f2bf(src[n * HID + k]);
}

// ---------------------------------------------------------------------------
// Layer-0 input projection: U0[t][b][n] = x[b][t] * Wih0[n] + bih0[n] + bhh0[n]
// ---------------------------------------------------------------------------
__global__ __launch_bounds__(256) void drnn_u0(
    const float* __restrict__ x, const float* __restrict__ Wih0,
    const float* __restrict__ bih0, const float* __restrict__ bhh0,
    unsigned short* __restrict__ U0)
{
    size_t i = (size_t)blockIdx.x * 256 + threadIdx.x;
    if (i >= (size_t)T_STEPS * BATCH * HID) return;
    int n = (int)(i & (HID - 1));
    size_t tb = i >> 7;
    int b = (int)(tb % BATCH);
    int t = (int)(tb / BATCH);
    float v = x[(size_t)b * T_STEPS + t] * Wih0[n] + bih0[n] + bhh0[n];
    U0[i] = f2bf(v);
}

// ---------------------------------------------------------------------------
// U GEMM, layers 1..7: Uout[row][n] = sum_k Hin[row][k]*Wih[n][k] + bias[n]
// 16 rows per workgroup; wave w owns N-tile [16w,16w+16). No LDS needed:
// B-operands come straight from the packed global image.
// ---------------------------------------------------------------------------
__global__ __launch_bounds__(256) void drnn_ugemm(
    const unsigned short* __restrict__ Hin,   // [T*B][128] bf16
    const unsigned short* __restrict__ wp,    // packed Wih (16384 bf16)
    const float* __restrict__ bih, const float* __restrict__ bhh,
    unsigned short* __restrict__ Uout)        // [T*B][128] bf16
{
    const int tid  = threadIdx.x;
    const int lane = tid & 31;
    const int wave = tid >> 5;
    const int grp  = lane >> 4;
    const int ln   = lane & 15;
    const int col  = wave * 16 + ln;

    v16bf breg[4];
#pragma unroll
    for (int s = 0; s < 4; ++s)
        breg[s] = *(const v16bf*)(wp + ((size_t)(wave * 4 + s) * 32 + lane) * 16);

    const int rowbase = blockIdx.x * 16;
    const float bias = bih[col] + bhh[col];
    v8f c;
#pragma unroll
    for (int v = 0; v < 8; ++v) c[v] = bias;

    const int off0 = grp * 8;       // ISA 16-bit A layout K-offset
    const int arow = rowbase + ln;
#pragma unroll
    for (int ks = 0; ks < 4; ++ks) {
        int kb = ks * 32;
        bf16x8 lo = *(const bf16x8*)(Hin + (size_t)arow * HID + kb + off0);
        bf16x8 hi = *(const bf16x8*)(Hin + (size_t)arow * HID + kb + off0 + 16);
        v16bf a = __builtin_shufflevector(lo, hi, 0, 1, 2, 3, 4, 5, 6, 7,
                                          8, 9, 10, 11, 12, 13, 14, 15);
        c = __builtin_amdgcn_wmma_f32_16x16x32_bf16(
                false, a, false, breg[ks], (short)0, c, false, false);
    }

#pragma unroll
    for (int v = 0; v < 8; ++v) {
        int m = v + grp * 8;        // C/D layout: VGPR v <-> row v (+8 upper)
        Uout[(size_t)(rowbase + m) * HID + col] = f2bf(c[v]);
    }
}

// ---------------------------------------------------------------------------
// Persistent dilated scan: each workgroup owns 16 folded rows (same j = t%rate,
// contiguous b), loops s = 0..n_dil-1 with t = s*rate + j.
//   h_t = tanh(U[t] + h_{t-rate} @ Whh^T)
// Packed Whh B-operands loaded ONCE into VGPRs; only the 4 KB h tile in LDS.
// ---------------------------------------------------------------------------
__global__ __launch_bounds__(256) void drnn_scan(
    const unsigned short* __restrict__ U,     // [T*B][128] bf16
    const unsigned short* __restrict__ wp,    // packed Whh (16384 bf16)
    unsigned short* __restrict__ Hout,        // [T*B][128] bf16
    int rate, int n_dil, int store_all)
{
    __shared__ unsigned short hA[16][HID];    // running h tile (A-operand)

    const int tid  = threadIdx.x;
    const int lane = tid & 31;
    const int wave = tid >> 5;
    const int grp  = lane >> 4;
    const int ln   = lane & 15;
    const int col  = wave * 16 + ln;

    for (int i = tid; i < 16 * HID; i += 256) ((unsigned short*)hA)[i] = 0;

    v16bf breg[4];
#pragma unroll
    for (int s = 0; s < 4; ++s)
        breg[s] = *(const v16bf*)(wp + ((size_t)(wave * 4 + s) * 32 + lane) * 16);

    __syncthreads();

    const int f     = blockIdx.x * 16;  // folded row base (j*B + b)
    const int j     = f / BATCH;        // all 16 rows share j (16 | 256)
    const int bbase = f % BATCH;
    const int off0  = grp * 8;
    const int row   = ln;

    for (int s = 0; s < n_dil; ++s) {
        int t = s * rate + j;
        if (t >= T_STEPS) break;        // padded tail: outputs dropped, no deps

        const unsigned short* Ut = U + ((size_t)t * BATCH) * HID;

        // prefetch next step's U tile while this step computes
        int tn = t + rate;
        if (tn < T_STEPS) {
            const unsigned short* Un = U + ((size_t)tn * BATCH) * HID;
            __builtin_prefetch(Un + (size_t)(bbase + grp * 8) * HID + col, 0, 0);
        }

        // C init from U (C/D layout: VGPR v -> row v + 8*grp, col)
        v8f c;
#pragma unroll
        for (int v = 0; v < 8; ++v) {
            int m = v + grp * 8;
            c[v] = bf2f(Ut[(size_t)(bbase + m) * HID + col]);
        }

        // h_{prev} @ Whh^T : 4 x wmma over K=128
#pragma unroll
        for (int ks = 0; ks < 4; ++ks) {
            int kb = ks * 32;
            bf16x8 lo = *(const bf16x8*)&hA[row][kb + off0];
            bf16x8 hi = *(const bf16x8*)&hA[row][kb + off0 + 16];
            v16bf a = __builtin_shufflevector(lo, hi, 0, 1, 2, 3, 4, 5, 6, 7,
                                              8, 9, 10, 11, 12, 13, 14, 15);
            c = __builtin_amdgcn_wmma_f32_16x16x32_bf16(
                    false, a, false, breg[ks], (short)0, c, false, false);
        }

        float hv[8];
#pragma unroll
        for (int v = 0; v < 8; ++v) hv[v] = fast_tanh(c[v]);

        __syncthreads();                // all A-reads of this step done
        const bool wr = store_all || (t == T_STEPS - 1);
        unsigned short* Ht = Hout + ((size_t)t * BATCH) * HID;
#pragma unroll
        for (int v = 0; v < 8; ++v) {
            int m = v + grp * 8;
            unsigned short hb = f2bf(hv[v]);
            hA[m][col] = hb;                                  // next-step A
            if (wr) Ht[(size_t)(bbase + m) * HID + col] = hb; // layer output
        }
        __syncthreads();                // h tile visible before next A-reads
    }
}

// ---------------------------------------------------------------------------
// Head: logits[b][c] = Hlast[T-1][b] @ W + bcls   (256 x 10, fp32 out)
// ---------------------------------------------------------------------------
__global__ __launch_bounds__(256) void drnn_head(
    const unsigned short* __restrict__ Hlast, const float* __restrict__ W,
    const float* __restrict__ bcls, float* __restrict__ out)
{
    int i = blockIdx.x * 256 + threadIdx.x;
    if (i >= BATCH * 10) return;
    int b = i / 10, cc = i % 10;
    const unsigned short* h =
        Hlast + ((size_t)((T_STEPS - 1) * BATCH + b)) * HID;
    float acc = bcls[cc];
#pragma unroll 4
    for (int k = 0; k < HID; ++k) acc += bf2f(h[k]) * W[k * 10 + cc];
    out[i] = acc;
}

extern "C" void kernel_launch(void* const* d_in, const int* in_sizes, int n_in,
                              void* d_out, int out_size, void* d_ws, size_t ws_size,
                              hipStream_t stream) {
    const float* x    = (const float*)d_in[0];
    const float* Wih0 = (const float*)d_in[1];
    const float* Whh0 = (const float*)d_in[2];
    const float* bih0 = (const float*)d_in[3];
    const float* bhh0 = (const float*)d_in[4];
    const float* Wih  = (const float*)d_in[5];   // (7,128,128)
    const float* Whh  = (const float*)d_in[6];   // (7,128,128)
    const float* bih  = (const float*)d_in[7];   // (7,128)
    const float* bhh  = (const float*)d_in[8];   // (7,128)
    const float* Wc   = (const float*)d_in[9];   // (128,10)
    const float* bc   = (const float*)d_in[10];  // (10,)

    const size_t TB = (size_t)T_STEPS * BATCH;   // 200704
    unsigned short* U    = (unsigned short*)d_ws;
    unsigned short* Ha   = U  + TB * HID;
    unsigned short* Hb   = Ha + TB * HID;
    unsigned short* whhp = Hb + TB * HID;        // 8 * 16384 bf16
    unsigned short* wihp = whhp + 8 * 16384;     // 7 * 16384 bf16

    static const int DIL[8] = {1, 2, 4, 8, 16, 32, 64, 128};

    // Pack all recurrent/input weight matrices into WMMA B-register images.
    drnn_pack<<<(15 * 16384 + 255) / 256, 256, 0, stream>>>(
        Whh0, Wih, Whh, whhp, wihp);

    // Layer 0: rank-1 input projection, then rate-1 scan.
    drnn_u0<<<(unsigned)((TB * HID + 255) / 256), 256, 0, stream>>>(
        x, Wih0, bih0, bhh0, U);
    drnn_scan<<<(1 * BATCH) / 16, 256, 0, stream>>>(
        U, whhp, Ha, 1, T_STEPS, 1);

    unsigned short* Hp = Ha;
    unsigned short* Hn = Hb;
    for (int l = 1; l < 8; ++l) {
        int r = DIL[l];
        drnn_ugemm<<<(unsigned)(TB / 16), 256, 0, stream>>>(
            Hp, wihp + (size_t)(l - 1) * 16384,
            bih + (size_t)(l - 1) * HID, bhh + (size_t)(l - 1) * HID, U);
        int ndil = (T_STEPS + r - 1) / r;
        drnn_scan<<<(r * BATCH) / 16, 256, 0, stream>>>(
            U, whhp + (size_t)l * 16384, Hn, r, ndil, (l < 7) ? 1 : 0);
        unsigned short* tmp = Hp; Hp = Hn; Hn = tmp;
    }

    drnn_head<<<(BATCH * 10 + 255) / 256, 256, 0, stream>>>(
        Hp, Wc, bc, (float*)d_out);
}
